// DDGCN_57707180589747
// MI455X (gfx1250) — matrix operand order repaired and checked by reading, not compiled, not run on hardware
//
#include <hip/hip_runtime.h>
#include <cmath>

// ---------------------------------------------------------------------------
// DD-GCN forward for MI455X (gfx1250, wave32, WMMA).
// All matmul-shaped einsums use v_wmma_f32_16x16x32_f16 (f16 in, f32 acc).
// Tiles are staged in *fragment-major* order so each lane loads its whole
// v16h operand with two ds_load_b128 / global_load_b128 (no u16 gather).
// Staging hoists bounds checks out of the unrolled loops: one branch per
// thread, then float4 loads + unconditional ds_store_b16 scatter.
// ---------------------------------------------------------------------------

typedef __attribute__((ext_vector_type(16))) _Float16 v16h;
typedef __attribute__((ext_vector_type(8)))  float    v8f;

namespace ddgcn {
constexpr int B = 128, T = 60, V0 = 22, V1 = 11, CIN = 3, H = 64, NBLK = 3;
constexpr float EPS = 1e-5f;
constexpr int P = T * V0;     // 1320 flattened (t,v) positions
constexpr int PPAD = 1344;    // 42 * 32 (K padding for skip GEMM)
constexpr int KT = PPAD / 32; // 42 K-tiles
constexpr int MT = PPAD / 16; // 84 M-tiles
}
using namespace ddgcn;

__device__ __forceinline__ float softsign_f(float z) { return z / (1.0f + fabsf(z)); }

// Fragment-major index helpers (CDNA5 ISA 7.12.2, wave32, 16x16x32 f16).
// A (16xK=32): lane = ((k>>3)&1)*16 + m ; element e = ((k>>4)<<3) | (k&7)
// B (K=32x16): lane = ((k>>3)&1)*16 + n ; element e = ((k>>4)<<3) | (k&7)
__device__ __forceinline__ int a_frag(int m, int k) {
  return (((((k >> 3) & 1) * 16 + m) << 4) | (((k >> 4) << 3) | (k & 7)));
}
__device__ __forceinline__ int b_frag(int k, int n) {
  return (((((k >> 3) & 1) * 16 + n) << 4) | (((k >> 4) << 3) | (k & 7)));
}

__device__ __forceinline__ v8f wmma_ab(v16h a, const _Float16* Bf, v8f acc) {
  v16h b = *(const v16h*)(Bf + (threadIdx.x & 31) * 16);
  return __builtin_amdgcn_wmma_f32_16x16x32_f16(false, a, false, b, (short)0, acc,
                                                false, false);
}

// ---------------------------------------------------------------------------
// Kernel 1: pre-mask skip adjacency into fragment-major f16 tiles.
// Tile (mt, kt) occupies 512 halfs; element (lane, e) maps to matrix (p, q):
//   p = mt*16 + (lane&15), q = kt*32 + k(lane, e).
// a[t,v,i,j] = adj * mask0[t,v,j] * (t!=i) * (v!=j)
// ---------------------------------------------------------------------------
__global__ void k_prep_skipA(const float* __restrict__ adj, const float* __restrict__ mask0,
                             _Float16* __restrict__ Ap) {
  int idx = blockIdx.x * blockDim.x + threadIdx.x;
  if (idx >= MT * KT * 512) return;
  int tile = idx >> 9, within = idx & 511;
  int lane = within >> 4, e = within & 15;
  int kt = tile % KT, mt = tile / KT;
  int m = lane & 15, half = lane >> 4;
  int i2 = e >> 1, bb = e & 1;
  int k = ((i2 >= 4) ? 16 : 0) + half * 8 + 2 * (i2 & 3) + bb;
  int p = mt * 16 + m;
  int q = kt * 32 + k;
  float val = 0.f;
  if (p < P && q < P) {
    int t = p / V0, v = p % V0, ii = q / V0, jj = q % V0;
    if (t != ii && v != jj)
      val = adj[(((size_t)t * V0 + v) * T + ii) * V0 + jj] *
            mask0[((size_t)t * V0 + v) * V0 + jj];
  }
  Ap[idx] = (_Float16)val;
}

// ---------------------------------------------------------------------------
// Kernel 2: skip GEMM. Y[n,p,c] = sum_q A[p,q] * X[n,q,c]   (P x PPAD x 64)
// block = 128 (4 waves); each wave owns one 16-row M tile; block shares a
// 32x64 B panel per K step. A fragments come straight from packed global.
// grid = (MT/4 = 21, B)
// ---------------------------------------------------------------------------
__global__ void k_skip_gemm(const _Float16* __restrict__ Ap, const float* __restrict__ X,
                            float* __restrict__ Y) {
  __shared__ __align__(32) _Float16 Bs[4 * 512];
  const int n = blockIdx.y;
  const int tid = threadIdx.x;
  const int w = tid >> 5, lane = tid & 31;
  const int mt = blockIdx.x * 4 + w;
  const int row = tid >> 2, cseg = (tid & 3) * 16;
  v8f acc0 = {}, acc1 = {}, acc2 = {}, acc3 = {};
  for (int ks = 0; ks < KT; ++ks) {
    {  // stage B panel: rows k=0..31 (q = ks*32+k) x 64 channels -> frag-major
      int q = ks * 32 + row;
      _Float16 hv[16];
      if (q < P) {  // uniform per thread: one branch, four b128 loads
        const float4* s4 = (const float4*)(X + ((size_t)n * P + q) * H + cseg);
        if (q + 32 < P) __builtin_prefetch((const float*)s4 + 32 * H, 0, 1);
#pragma unroll
        for (int j = 0; j < 4; ++j) {
          float4 f = s4[j];
          hv[4 * j + 0] = (_Float16)f.x;
          hv[4 * j + 1] = (_Float16)f.y;
          hv[4 * j + 2] = (_Float16)f.z;
          hv[4 * j + 3] = (_Float16)f.w;
        }
      } else {
#pragma unroll
        for (int c = 0; c < 16; ++c) hv[c] = (_Float16)0.f;
      }
#pragma unroll
      for (int c = 0; c < 16; ++c) {
        int ng = cseg + c;
        Bs[(ng >> 4) * 512 + b_frag(row, ng & 15)] = hv[c];
      }
    }
    __syncthreads();
    v16h a = *(const v16h*)(Ap + (((size_t)mt * KT + ks) << 9) + lane * 16);
    acc0 = wmma_ab(a, Bs + 0 * 512, acc0);
    acc1 = wmma_ab(a, Bs + 1 * 512, acc1);
    acc2 = wmma_ab(a, Bs + 2 * 512, acc2);
    acc3 = wmma_ab(a, Bs + 3 * 512, acc3);
    __syncthreads();
  }
  const int half = lane >> 4, nn = lane & 15;
  const int m0 = mt * 16;
#pragma unroll
  for (int r = 0; r < 8; ++r) {
    int mm = m0 + half * 8 + r;
    if (mm < P) {
      float* dst = Y + ((size_t)n * P + mm) * H + nn;
      dst[0] = acc0[r];
      dst[16] = acc1[r];
      dst[32] = acc2[r];
      dst[48] = acc3[r];
    }
  }
}

// ---------------------------------------------------------------------------
// Kernel 3: channel mix. Out[m,v,n] = sum_k Ain[m,v,k] * W[v,k,n] + bias[n]
// block = 128 (4 waves, one M tile each, shared weight panel).
// grid = (B*T/64 = 120, 1, V).  ntiles = ceil(N/16) in {1, 4}.
// ---------------------------------------------------------------------------
__global__ void k_chanmix(const float* __restrict__ Ain, const float* __restrict__ W,
                          const float* __restrict__ bias, float* __restrict__ Out,
                          int K, int N, int V) {
  __shared__ __align__(32) _Float16 As[4 * 512];
  __shared__ __align__(32) _Float16 Bs[4 * 512];
  const int v = blockIdx.z;
  const int tid = threadIdx.x;
  const int w = tid >> 5, lane = tid & 31;
  const int m0 = (blockIdx.x * 4 + w) * 16;
  const int ntiles = (N + 15) >> 4;
  v8f acc0 = {}, acc1 = {}, acc2 = {}, acc3 = {};
  const int ksteps = (K + 31) / 32;
  for (int ks = 0; ks < ksteps; ++ks) {
    int k0 = ks * 32;
    {  // stage this wave's A tile (frag-major); rows contiguous in k
      int m = lane >> 1, c0 = (lane & 1) * 16;
      const float* srow = Ain + ((size_t)(m0 + m) * V + v) * K + k0 + c0;
      _Float16 hv[16];
      if (k0 + c0 + 16 <= K) {  // full 16-wide segment: vector loads
        const float4* s4 = (const float4*)srow;
#pragma unroll
        for (int j = 0; j < 4; ++j) {
          float4 f = s4[j];
          hv[4 * j + 0] = (_Float16)f.x;
          hv[4 * j + 1] = (_Float16)f.y;
          hv[4 * j + 2] = (_Float16)f.z;
          hv[4 * j + 3] = (_Float16)f.w;
        }
      } else {
#pragma unroll
        for (int c = 0; c < 16; ++c)
          hv[c] = (_Float16)((k0 + c0 + c < K) ? srow[c] : 0.f);
      }
#pragma unroll
      for (int c = 0; c < 16; ++c) As[w * 512 + a_frag(m, c0 + c)] = hv[c];
    }
    {  // stage weight panel cooperatively (frag-major); rows contiguous in n
      int row = tid >> 2, cseg = (tid & 3) * 16;
      int k = k0 + row;
      _Float16 hv[16];
      if (k < K && cseg + 16 <= N) {
        const float4* s4 = (const float4*)(W + ((size_t)v * K + k) * N + cseg);
#pragma unroll
        for (int j = 0; j < 4; ++j) {
          float4 f = s4[j];
          hv[4 * j + 0] = (_Float16)f.x;
          hv[4 * j + 1] = (_Float16)f.y;
          hv[4 * j + 2] = (_Float16)f.z;
          hv[4 * j + 3] = (_Float16)f.w;
        }
      } else {
#pragma unroll
        for (int c = 0; c < 16; ++c)
          hv[c] = (_Float16)((k < K && cseg + c < N) ? W[((size_t)v * K + k) * N + cseg + c]
                                                     : 0.f);
      }
      if (cseg < ntiles * 16) {
#pragma unroll
        for (int c = 0; c < 16; ++c) {
          int ng = cseg + c;
          Bs[(ng >> 4) * 512 + b_frag(row, ng & 15)] = hv[c];
        }
      }
    }
    __syncthreads();
    v16h a = *(const v16h*)(As + w * 512 + lane * 16);
    acc0 = wmma_ab(a, Bs + 0 * 512, acc0);
    if (ntiles > 1) {
      acc1 = wmma_ab(a, Bs + 1 * 512, acc1);
      acc2 = wmma_ab(a, Bs + 2 * 512, acc2);
      acc3 = wmma_ab(a, Bs + 3 * 512, acc3);
    }
    __syncthreads();
  }
  const int half = lane >> 4, nn0 = lane & 15;
#pragma unroll
  for (int nt = 0; nt < 4; ++nt) {
    if (nt < ntiles) {
      int nn = nt * 16 + nn0;
      if (nn < N) {
        float bv = bias[nn];
        v8f acc = (nt == 0) ? acc0 : (nt == 1) ? acc1 : (nt == 2) ? acc2 : acc3;
#pragma unroll
        for (int r = 0; r < 8; ++r) {
          int mm = m0 + half * 8 + r;  // B*T divisible by 64 -> in range
          Out[((size_t)mm * V + v) * N + nn] = acc[r] + bv;
        }
      }
    }
  }
}

// ---------------------------------------------------------------------------
// Kernel 4: spatial aggregation per (n,t): Out[v,c] = sum_i A[v,i] * X[i,c]
// mode 0 (static-nn): A = dinv[v] * s[t,v,i] * mask[t,v,i]
// mode 1 (dynamic):   A = s*mask + softsign((xm[v]-xm[i])*mask) * invsqd
// grid = (ceil(V/16), ceil(C/16), B*T), block = 32. Writes Out.
// ---------------------------------------------------------------------------
__global__ void k_agg_s(const float* __restrict__ X, const float* __restrict__ Sp,
                        const float* __restrict__ Mask, const float* __restrict__ Xm,
                        float* __restrict__ Out, int V, int C, int mode, float invsqd) {
  __shared__ __align__(32) _Float16 As[512];
  __shared__ __align__(32) _Float16 Bs[512];
  const int m0 = blockIdx.x * 16;
  const int n0 = blockIdx.y * 16;
  const int nt = blockIdx.z;  // n*T + t
  const int t = nt % T;
  const int tid = threadIdx.x;
  {  // stage adjacency tile (frag-major), computed on the fly
    int r = tid >> 1, c0 = (tid & 1) * 16;
    int vr = m0 + r;
#pragma unroll
    for (int c = 0; c < 16; ++c) {
      int i = c0 + c;
      float a = 0.f;
      if (vr < V && i < V) {
        float mk = Mask[((size_t)t * V + vr) * V + i];
        float sv = Sp[((size_t)t * V + vr) * V + i] * mk;
        if (mode == 0) {
          float d = 0.f;
          for (int u = 0; u < V; ++u) d += Mask[(size_t)vr * V + u];  // mask[0] row sum
          a = (d > 0.f) ? sv / d : 0.f;
        } else {
          float diff = Xm[(size_t)nt * V + vr] - Xm[(size_t)nt * V + i];
          a = sv + softsign_f(diff * mk) * invsqd;
        }
      }
      As[a_frag(r, i)] = (_Float16)a;
    }
  }
  {  // stage X tile (frag-major); rows contiguous in c
    int i = tid;
    _Float16 hv[16];
    if (i < V && n0 + 16 <= C) {
      const float4* s4 = (const float4*)(X + ((size_t)nt * V + i) * C + n0);
#pragma unroll
      for (int j = 0; j < 4; ++j) {
        float4 f = s4[j];
        hv[4 * j + 0] = (_Float16)f.x;
        hv[4 * j + 1] = (_Float16)f.y;
        hv[4 * j + 2] = (_Float16)f.z;
        hv[4 * j + 3] = (_Float16)f.w;
      }
    } else {
#pragma unroll
      for (int c = 0; c < 16; ++c)
        hv[c] = (_Float16)((i < V && n0 + c < C) ? X[((size_t)nt * V + i) * C + n0 + c] : 0.f);
    }
#pragma unroll
    for (int c = 0; c < 16; ++c) Bs[b_frag(i, c)] = hv[c];
  }
  __syncthreads();
  v16h a = *(const v16h*)(As + tid * 16);
  v8f acc = {};
  acc = wmma_ab(a, Bs, acc);
  const int half = tid >> 4;
  int cc = n0 + (tid & 15);
  if (cc < C) {
#pragma unroll
    for (int r = 0; r < 8; ++r) {
      int vv = m0 + half * 8 + r;
      if (vv < V) Out[((size_t)nt * V + vv) * C + cc] = acc[r];
    }
  }
}

// ---------------------------------------------------------------------------
// Kernel 5: temporal aggregation per (n,v): Out[t,c] += sum_j A[t,j] * X[j,c]
// A[t,j] = Tp[v,t,j] (+ softsign(xm[n,t,v]-xm[n,j,v])*invsqd when dynamic)
// grid = (4, ceil(C/16), B*V), block = 32. K padded 60 -> 64 (2 steps).
// ---------------------------------------------------------------------------
__global__ void k_agg_t(const float* __restrict__ X, const float* __restrict__ Tp,
                        const float* __restrict__ Xm, float* __restrict__ Out,
                        int V, int C, int dynamic, float invsqd) {
  __shared__ __align__(32) _Float16 As[512];
  __shared__ __align__(32) _Float16 Bs[512];
  const int m0 = blockIdx.x * 16;
  const int n0 = blockIdx.y * 16;
  const int nv = blockIdx.z;
  const int n = nv / V, v = nv % V;
  const int tid = threadIdx.x;
  v8f acc = {};
  for (int k0 = 0; k0 < 64; k0 += 32) {
    {
      int r = tid >> 1, c0 = (tid & 1) * 16;
      int trow = m0 + r;
#pragma unroll
      for (int c = 0; c < 16; ++c) {
        int kk = c0 + c, j = k0 + kk;
        float a = 0.f;
        if (trow < T && j < T) {
          a = Tp[((size_t)v * T + trow) * T + j];
          if (dynamic) {
            float diff = Xm[((size_t)n * T + trow) * V + v] - Xm[((size_t)n * T + j) * V + v];
            a += softsign_f(diff) * invsqd;
          }
        }
        As[a_frag(r, kk)] = (_Float16)a;
      }
    }
    {
      int j = k0 + tid;
      _Float16 hv[16];
      if (j < T && n0 + 16 <= C) {
        const float4* s4 = (const float4*)(X + ((size_t)(n * T + j) * V + v) * C + n0);
#pragma unroll
        for (int jj = 0; jj < 4; ++jj) {
          float4 f = s4[jj];
          hv[4 * jj + 0] = (_Float16)f.x;
          hv[4 * jj + 1] = (_Float16)f.y;
          hv[4 * jj + 2] = (_Float16)f.z;
          hv[4 * jj + 3] = (_Float16)f.w;
        }
      } else {
#pragma unroll
        for (int c = 0; c < 16; ++c)
          hv[c] = (_Float16)((j < T && n0 + c < C)
                                 ? X[((size_t)(n * T + j) * V + v) * C + n0 + c]
                                 : 0.f);
      }
#pragma unroll
      for (int c = 0; c < 16; ++c) Bs[b_frag(tid, c)] = hv[c];
    }
    __syncthreads();
    v16h a = *(const v16h*)(As + tid * 16);
    acc = wmma_ab(a, Bs, acc);
    __syncthreads();
  }
  const int half = tid >> 4;
  int cc = n0 + (tid & 15);
  if (cc < C) {
#pragma unroll
    for (int r = 0; r < 8; ++r) {
      int trow = m0 + half * 8 + r;
      if (trow < T) Out[((size_t)(n * T + trow) * V + v) * C + cc] += acc[r];
    }
  }
}

// ---------------------------------------------------------------------------
// Elementwise / reduction kernels
// ---------------------------------------------------------------------------
__global__ void k_mean_c(const float* __restrict__ X, float* __restrict__ Xm, int C, int total) {
  int idx = blockIdx.x * blockDim.x + threadIdx.x;
  if (idx >= total) return;
  const float* p = X + (size_t)idx * C;
  float s = 0.f;
  for (int c = 0; c < C; ++c) s += p[c];
  Xm[idx] = s / (float)C;
}

__global__ void k_zero(float* p, int n) {
  int i = blockIdx.x * blockDim.x + threadIdx.x;
  if (i < n) p[i] = 0.f;
}

// per-channel sum / sumsq over rows (C fixed at 64). block=256 (4 row-groups x 64 ch)
__global__ void k_bn_stats(const float* __restrict__ X, float* __restrict__ stats, int rows) {
  __shared__ float sm[512];
  const int tid = threadIdx.x;
  const int c = tid & 63, g = tid >> 6;
  float s1 = 0.f, s2 = 0.f;
  for (int r = blockIdx.x * 4 + g; r < rows; r += gridDim.x * 4) {
    float x = X[(size_t)r * 64 + c];
    s1 += x;
    s2 += x * x;
  }
  sm[tid] = s1;
  sm[256 + tid] = s2;
  __syncthreads();
  if (tid < 64) {
    float a1 = sm[tid] + sm[tid + 64] + sm[tid + 128] + sm[tid + 192];
    float a2 = sm[256 + tid] + sm[256 + tid + 64] + sm[256 + tid + 128] + sm[256 + tid + 192];
    atomicAdd(&stats[c], a1);
    atomicAdd(&stats[64 + c], a2);
  }
}

// y = tanh((x - mean) * rsqrt(var+eps) * gamma + beta) [+ res]
__global__ void k_bn_tanh(const float* __restrict__ X, const float* __restrict__ stats,
                          const float* __restrict__ gamma, const float* __restrict__ beta,
                          const float* __restrict__ res, float* __restrict__ Out,
                          int total, float invN) {
  int idx = blockIdx.x * blockDim.x + threadIdx.x;
  if (idx >= total) return;
  int c = idx & 63;
  float m = stats[c] * invN;
  float var = stats[64 + c] * invN - m * m;
  float rs = rsqrtf(var + EPS);
  float y = tanhf((X[idx] - m) * rs * gamma[c] + beta[c]);
  if (res) y += res[idx];
  Out[idx] = y;
}

// Out[bt, nO, c] (+)= sum_v X[bt, v, c] * Tr[v, nO]
__global__ void k_pool(const float* __restrict__ X, const float* __restrict__ Tr,
                       float* __restrict__ Out, int Vin, int Vout, int C, int accum,
                       int total) {
  int idx = blockIdx.x * blockDim.x + threadIdx.x;
  if (idx >= total) return;
  int c = idx % C;
  int rest = idx / C;
  int nO = rest % Vout;
  int bt = rest / Vout;
  float s = 0.f;
  for (int v = 0; v < Vin; ++v) s += X[((size_t)bt * Vin + v) * C + c] * Tr[v * Vout + nO];
  if (accum) Out[idx] += s;
  else Out[idx] = s;
}

__global__ void k_add(const float* __restrict__ A, const float* __restrict__ Bp,
                      float* __restrict__ O, int n) {
  int i = blockIdx.x * blockDim.x + threadIdx.x;
  if (i < n) O[i] = A[i] + Bp[i];
}

// ---------------------------------------------------------------------------
// Host orchestration
// Input flattening (setup_inputs dict order, nested depth-first, lists in order):
//  0:x 1:transform1 2-6:inv_t[0..4]
//  7-12: enc0 {w,s,t,b,gamma,beta}   13-18: enc1 {...}
//  19+i*12: blocks0[i] {gc1.w,gc1.s,gc1.t,gc1.b, gc2.w,gc2.s,gc2.t,gc2.b, g1,b1n,g2,b2n}
//  55+i*12: blocks1[i]
//  91-94: de0 {w,s,t,b}  95-98: de1  99: skip_adj  100: mask0  101: mask1
// ---------------------------------------------------------------------------
extern "C" void kernel_launch(void* const* d_in, const int* in_sizes, int n_in,
                              void* d_out, int out_size, void* d_ws, size_t ws_size,
                              hipStream_t stream) {
  (void)in_sizes; (void)n_in; (void)out_size; (void)ws_size;
  auto F = [&](int i) { return (const float*)d_in[i]; };
  const float* x = F(0);
  const float* transform1 = F(1);
  const float* inv_t[5] = {F(2), F(3), F(4), F(5), F(6)};
  const float* mask0 = F(100);
  const float* mask1 = F(101);

  // workspace bump allocator (256B aligned)
  char* base = (char*)d_ws;
  size_t off = 0;
  auto alloc = [&](size_t bytes) -> void* {
    void* p = base + off;
    off += (bytes + 255) & ~(size_t)255;
    return p;
  };
  const size_t A0 = (size_t)B * T * V0 * H;
  const size_t A1 = (size_t)B * T * V1 * H;
  _Float16* skipA = (_Float16*)alloc((size_t)MT * KT * 512 * sizeof(_Float16));
  float* y0 = (float*)alloc(A0 * 4);
  float* h0 = (float*)alloc(A0 * 4);
  float* ag0 = (float*)alloc(A0 * 4);
  float* yskip = (float*)alloc(A0 * 4);
  float* y1 = (float*)alloc(A1 * 4);
  float* h1 = (float*)alloc(A1 * 4);
  float* ag1 = (float*)alloc(A1 * 4);
  float* xm0 = (float*)alloc((size_t)B * T * V0 * 4);
  float* xm1 = (float*)alloc((size_t)B * T * V1 * 4);
  float* x1b = (float*)alloc((size_t)B * T * V1 * CIN * 4);
  float* d0 = (float*)alloc((size_t)B * T * V0 * CIN * 4);
  float* d1 = (float*)alloc((size_t)B * T * V1 * CIN * 4);
  float* stats = (float*)alloc(128 * 4);

  // ---- helpers ----
  auto launch_bn = [&](float* xb, const float* g, const float* bt, const float* res,
                       float* out, int V) {
    int total = B * T * V * H;
    k_zero<<<1, 128, 0, stream>>>(stats, 128);
    k_bn_stats<<<256, 256, 0, stream>>>(xb, stats, B * T * V);
    k_bn_tanh<<<(total + 255) / 256, 256, 0, stream>>>(xb, stats, g, bt, res, out, total,
                                                       1.0f / (float)(B * T * V));
  };
  auto launch_gc = [&](const float* src, float* dst, float* ag, float* xm,
                       const float* sP, const float* tP, const float* wP, const float* bP,
                       const float* mask, int V, int Cin, int Cout) {
    int rows = B * T * V;
    k_mean_c<<<(rows + 255) / 256, 256, 0, stream>>>(src, xm, Cin, rows);
    float invsqd = 1.0f / sqrtf((float)Cout);
    dim3 gS((V + 15) / 16, (Cin + 15) / 16, B * T);
    k_agg_s<<<gS, 32, 0, stream>>>(src, sP, mask, xm, ag, V, Cin, 1, invsqd);
    dim3 gT(4, (Cin + 15) / 16, B * V);
    k_agg_t<<<gT, 32, 0, stream>>>(src, tP, xm, ag, V, Cin, 1, invsqd);
    dim3 gC((B * T) / 64, 1, V);
    k_chanmix<<<gC, 128, 0, stream>>>(ag, wP, bP, dst, Cin, Cout, V);
  };
  auto launch_enc = [&](const float* src, float* dst, float* ag, const float* sP,
                        const float* tP, const float* wP, const float* bP,
                        const float* gam, const float* bet, const float* mask, int V) {
    dim3 gS((V + 15) / 16, 1, B * T);  // Cin=3 -> one N tile
    k_agg_s<<<gS, 32, 0, stream>>>(src, sP, mask, nullptr, ag, V, CIN, 0, 0.f);
    dim3 gT(4, 1, B * V);
    k_agg_t<<<gT, 32, 0, stream>>>(src, tP, nullptr, ag, V, CIN, 0, 0.f);
    dim3 gC((B * T) / 64, 1, V);
    k_chanmix<<<gC, 128, 0, stream>>>(ag, wP, bP, dst, CIN, H, V);
    launch_bn(dst, gam, bet, nullptr, dst, V);
  };
  auto launch_block = [&](float* y, float* h, float* ag, float* xm, const float* mask,
                          int V, int pb) {
    launch_gc(y, h, ag, xm, F(pb + 1), F(pb + 2), F(pb + 0), F(pb + 3), mask, V, H, H);
    launch_bn(h, F(pb + 8), F(pb + 9), nullptr, h, V);
    launch_gc(h, h, ag, xm, F(pb + 5), F(pb + 6), F(pb + 4), F(pb + 7), mask, V, H, H);
    launch_bn(h, F(pb + 10), F(pb + 11), y, y, V);  // + residual, back into y
  };

  // ---- pipeline ----
  {  // masked skip adjacency -> packed fragment-major f16 tiles
    int tot = MT * KT * 512;
    k_prep_skipA<<<(tot + 255) / 256, 256, 0, stream>>>(F(99), mask0, skipA);
  }
  {  // x1 = pool(x, transform1)
    int tot = B * T * V1 * CIN;
    k_pool<<<(tot + 255) / 256, 256, 0, stream>>>(x, transform1, x1b, V0, V1, CIN, 0, tot);
  }

  // encoders (static-nn mode + BN/tanh)
  launch_enc(x, y0, ag0, F(8), F(9), F(7), F(10), F(11), F(12), mask0, V0);
  launch_enc(x1b, y1, ag1, F(14), F(15), F(13), F(16), F(17), F(18), mask1, V1);

  {  // y_skip = skip(y0)  -- the big WMMA GEMM (64x64 macro tiles, 4 waves)
    dim3 g(MT / 4, B);
    k_skip_gemm<<<g, 128, 0, stream>>>(skipA, y0, yskip);
  }
  {  // y0 += pool(y1, inv_t[0])
    int tot = (int)A0;
    k_pool<<<(tot + 255) / 256, 256, 0, stream>>>(y1, inv_t[0], y0, V1, V0, H, 1, tot);
  }

  for (int i = 0; i < NBLK; ++i) {
    launch_block(y0, h0, ag0, xm0, mask0, V0, 19 + i * 12);
    launch_block(y1, h1, ag1, xm1, mask1, V1, 55 + i * 12);
    int tot = (int)A0;
    k_pool<<<(tot + 255) / 256, 256, 0, stream>>>(y1, inv_t[i + 1], y0, V1, V0, H, 1, tot);
  }

  {  // y0 += y_skip
    int tot = (int)A0;
    k_add<<<(tot + 255) / 256, 256, 0, stream>>>(y0, yskip, y0, tot);
  }

  // decoders (dynamic GC, H -> CIN, no BN)
  launch_gc(y0, d0, ag0, xm0, F(92), F(93), F(91), F(94), mask0, V0, H, CIN);
  launch_gc(y1, d1, ag1, xm1, F(96), F(97), F(95), F(98), mask1, V1, H, CIN);
  {
    int tot = B * T * V0 * CIN;
    k_pool<<<(tot + 255) / 256, 256, 0, stream>>>(d1, inv_t[4], d0, V1, V0, CIN, 1, tot);
    k_add<<<(tot + 255) / 256, 256, 0, stream>>>(d0, x, (float*)d_out, tot);
  }
}
